// BitNet_24163486007706
// MI455X (gfx1250) — 16-bit and 8-bit matrix operand order repaired and checked by reading, compiled, not run on hardware
//
#include <hip/hip_runtime.h>
#include <cstdint>

#define VOCAB  32000
#define NEMBD  1024
#define NHEAD  16
#define HD     64
#define NLAYER 4
#define BB     2
#define TT     1024
#define MM     (BB*TT)
#define FFND   4096
#define EPSV   1e-6f

typedef __attribute__((ext_vector_type(8)))  int    v8i;
typedef __attribute__((ext_vector_type(8)))  float  v8f;
typedef __attribute__((ext_vector_type(16))) __bf16 v16bf;

struct B32i2 { int2 a, b, c, d; };
struct B32i4 { int4 a, b; };

#if __has_builtin(__builtin_amdgcn_global_load_async_to_lds_b128)
#define HAVE_ASYNC 1
#else
#define HAVE_ASYNC 0
#endif

typedef int gv4i __attribute__((vector_size(16)));
typedef __attribute__((address_space(1))) gv4i* gas1_v4i;
typedef __attribute__((address_space(3))) gv4i* gas3_v4i;

__device__ __forceinline__ void async_cp16(void* lds_dst, const void* gsrc) {
#if HAVE_ASYNC
  __builtin_amdgcn_global_load_async_to_lds_b128(
      (gas1_v4i)(void*)gsrc, (gas3_v4i)lds_dst, 0, 0);
#else
  *(int4*)lds_dst = *(const int4*)gsrc;
#endif
}
__device__ __forceinline__ void wait_async0() {
#if HAVE_ASYNC
#if __has_builtin(__builtin_amdgcn_s_wait_asynccnt)
  __builtin_amdgcn_s_wait_asynccnt(0);
#else
  asm volatile("s_wait_asynccnt 0x0" ::: "memory");
#endif
#endif
}

__device__ __forceinline__ unsigned short f2bf(float f) {
  unsigned int u = __builtin_bit_cast(unsigned int, f);
  u += 0x7FFFu + ((u >> 16) & 1u);
  return (unsigned short)(u >> 16);
}
__device__ __forceinline__ unsigned int pack2bf(float a, float b) {
  return (unsigned int)f2bf(a) | ((unsigned int)f2bf(b) << 16);
}

__global__ void k_embed(const int* __restrict__ idx, const float* __restrict__ tok,
                        const float* __restrict__ pos, float* __restrict__ x) {
  int row = blockIdx.x;
  int t = row % TT;
  int token = idx[row];
  int c = threadIdx.x * 4;
  float4 a = *(const float4*)(tok + (size_t)token * NEMBD + c);
  float4 p = *(const float4*)(pos + (size_t)t * NEMBD + c);
  float4 o = {a.x + p.x, a.y + p.y, a.z + p.z, a.w + p.w};
  *(float4*)(x + (size_t)row * NEMBD + c) = o;
}

__global__ void k_absmean_part(const float* __restrict__ w, float* __restrict__ part, int n) {
  float s = 0.f;
  for (int i = blockIdx.x * blockDim.x + threadIdx.x; i < n; i += gridDim.x * blockDim.x)
    s += fabsf(w[i]);
  __shared__ float red[256];
  red[threadIdx.x] = s; __syncthreads();
  for (int k = 128; k > 0; k >>= 1) {
    if (threadIdx.x < k) red[threadIdx.x] += red[threadIdx.x + k];
    __syncthreads();
  }
  if (threadIdx.x == 0) part[blockIdx.x] = red[0];
}

__global__ void k_absmean_fin(const float* __restrict__ part, float* __restrict__ scale, float invn) {
  __shared__ float red[128];
  red[threadIdx.x] = part[threadIdx.x]; __syncthreads();
  for (int k = 64; k > 0; k >>= 1) {
    if (threadIdx.x < k) red[threadIdx.x] += red[threadIdx.x + k];
    __syncthreads();
  }
  if (threadIdx.x == 0) scale[0] = fmaxf(red[0] * invn, 1e-5f);
}

__global__ void k_wquant(const float* __restrict__ w, const float* __restrict__ scale,
                         int8_t* __restrict__ qw, int n) {
  float inv = 1.f / scale[0];
  for (int i = blockIdx.x * blockDim.x + threadIdx.x; i < n; i += gridDim.x * blockDim.x) {
    float q = rintf(w[i] * inv);
    q = fmaxf(-1.f, fminf(1.f, q));
    qw[i] = (int8_t)(int)q;
  }
}

__global__ void k_rmsnorm(const float* __restrict__ x, const float* __restrict__ w,
                          float* __restrict__ out, int K) {
  int row = blockIdx.x, tid = threadIdx.x;
  const float* xr = x + (size_t)row * K;
  float ss = 0.f;
  for (int c = tid * 4; c < K; c += 1024) {
    float4 v = *(const float4*)(xr + c);
    ss += v.x*v.x + v.y*v.y + v.z*v.z + v.w*v.w;
  }
  __shared__ float red[256];
  red[tid] = ss; __syncthreads();
  for (int k = 128; k > 0; k >>= 1) { if (tid < k) red[tid] += red[tid + k]; __syncthreads(); }
  float rstd = rsqrtf(red[0] / K + EPSV);
  for (int c = tid * 4; c < K; c += 1024) {
    float4 v = *(const float4*)(xr + c);
    float4 g = *(const float4*)(w + c);
    float4 o = {v.x*g.x*rstd, v.y*g.y*rstd, v.z*g.z*rstd, v.w*g.w*rstd};
    *(float4*)(out + (size_t)row * K + c) = o;
  }
}

__global__ void k_rq(const float* __restrict__ x, const float* __restrict__ nw,
                     int8_t* __restrict__ q, float* __restrict__ ascale, int K) {
  int row = blockIdx.x, tid = threadIdx.x;
  const float* xr = x + (size_t)row * K;
  int nch = K >> 10;
  float4 xv[4], wv[4];
  float ss = 0.f;
  for (int c = 0; c < nch; ++c) {
    xv[c] = *(const float4*)(xr + c*1024 + tid*4);
    wv[c] = *(const float4*)(nw + c*1024 + tid*4);
    ss += xv[c].x*xv[c].x + xv[c].y*xv[c].y + xv[c].z*xv[c].z + xv[c].w*xv[c].w;
  }
  __shared__ float red[256];
  red[tid] = ss; __syncthreads();
  for (int k = 128; k > 0; k >>= 1) { if (tid < k) red[tid] += red[tid + k]; __syncthreads(); }
  float rstd = rsqrtf(red[0] / K + EPSV);
  __syncthreads();
  float y[16];
  float amax = 0.f;
  for (int c = 0; c < nch; ++c) {
    y[c*4+0] = xv[c].x * wv[c].x * rstd;
    y[c*4+1] = xv[c].y * wv[c].y * rstd;
    y[c*4+2] = xv[c].z * wv[c].z * rstd;
    y[c*4+3] = xv[c].w * wv[c].w * rstd;
    amax = fmaxf(amax, fmaxf(fmaxf(fabsf(y[c*4+0]), fabsf(y[c*4+1])),
                             fmaxf(fabsf(y[c*4+2]), fabsf(y[c*4+3]))));
  }
  red[tid] = amax; __syncthreads();
  for (int k = 128; k > 0; k >>= 1) { if (tid < k) red[tid] = fmaxf(red[tid], red[tid + k]); __syncthreads(); }
  float s = 127.f / fmaxf(red[0], 1e-5f);
  for (int c = 0; c < nch; ++c) {
    unsigned int p = 0;
    #pragma unroll
    for (int e = 0; e < 4; ++e) {
      float qv = rintf(y[c*4+e] * s);
      qv = fmaxf(-128.f, fminf(127.f, qv));
      p |= ((unsigned int)(int)qv & 0xFFu) << (8 * e);
    }
    *(unsigned int*)(q + (size_t)row * K + c*1024 + tid*4) = p;
  }
  if (tid == 0) ascale[row] = s;
}

__device__ __forceinline__ void stage_i8(const int8_t* __restrict__ qa,
                                         const int8_t* __restrict__ qw,
                                         int8_t* Asb, int8_t* Bsb,
                                         int m0, int n0, int K, int k0, int tid) {
  #pragma unroll
  for (int i = 0; i < 2; ++i) {
    int idx = tid + i * 256;
    int r = idx >> 2, c = (idx & 3) << 4;
    async_cp16(Asb + r * 64 + c, qa + (size_t)(m0 + r) * K + k0 + c);
  }
  {
    int r = tid >> 2, c = (tid & 3) << 4;
    async_cp16(Bsb + r * 64 + c, qw + (size_t)(n0 + r) * K + k0 + c);
  }
}

__global__ void __launch_bounds__(256) k_gemm_i8(
    const int8_t* __restrict__ qa, const float* __restrict__ ascale,
    const int8_t* __restrict__ qw, const float* __restrict__ wscale,
    const float* __restrict__ res, float* __restrict__ out,
    int N, int K, int mode) {
  __shared__ __align__(16) int8_t As[2][128 * 64];
  __shared__ __align__(16) int8_t Bs[2][64 * 64];
  const int tid = threadIdx.x;
  const int m0 = blockIdx.y * 128;
  const int n0 = blockIdx.x * 64;
  const int lane = tid & 31, wave = tid >> 5;
  const int wm = (wave & 3) * 32, wn = (wave >> 2) * 32;
  const int l16 = lane & 15, lh = lane >> 4;
  const int KS = K >> 6;

  v8i acc[2][2] = {};
  stage_i8(qa, qw, As[0], Bs[0], m0, n0, K, 0, tid);
  wait_async0();
  __syncthreads();

  for (int ks = 0; ks < KS; ++ks) {
    int cur = ks & 1;
    if (ks + 1 < KS)
      stage_i8(qa, qw, As[cur ^ 1], Bs[cur ^ 1], m0, n0, K, (ks + 1) << 6, tid);

    const int2* A2 = (const int2*)As[cur];
    const int4* B4 = (const int4*)Bs[cur];
    B32i2 af[2];
    #pragma unroll
    for (int mi = 0; mi < 2; ++mi) {
      int r = wm + mi * 16 + l16;
      int base = r * 8 + 2 * lh;
      af[mi].a = A2[base];     af[mi].b = A2[base + 1];
      af[mi].c = A2[base + 4]; af[mi].d = A2[base + 5];
    }
    B32i4 bfg[2];
    #pragma unroll
    for (int ni = 0; ni < 2; ++ni) {
      int n = wn + ni * 16 + l16;
      bfg[ni].a = B4[n * 4 + lh];
      bfg[ni].b = B4[n * 4 + 2 + lh];
    }
    #pragma unroll
    for (int mi = 0; mi < 2; ++mi) {
      v8i a = __builtin_bit_cast(v8i, af[mi]);
      #pragma unroll
      for (int ni = 0; ni < 2; ++ni) {
        v8i b = __builtin_bit_cast(v8i, bfg[ni]);
        acc[mi][ni] = __builtin_amdgcn_wmma_i32_16x16x64_iu8(
            true, a, true, b, acc[mi][ni], false, false);
      }
    }
    wait_async0();
    __syncthreads();
  }

  float sw = wscale[0];
  #pragma unroll
  for (int mi = 0; mi < 2; ++mi)
    #pragma unroll
    for (int ni = 0; ni < 2; ++ni) {
      int col = n0 + wn + ni * 16 + l16;
      #pragma unroll
      for (int j = 0; j < 8; ++j) {
        int row = m0 + wm + mi * 16 + lh * 8 + j;
        float val = (float)acc[mi][ni][j] * sw / ascale[row];
        if (mode == 1) val = 0.5f * val * (1.f + erff(val * 0.70710678f));
        if (res) val += res[(size_t)row * N + col];
        out[(size_t)row * N + col] = val;
      }
    }
}

__global__ void __launch_bounds__(256) k_gemm_bf16(
    const float* __restrict__ A, const float* __restrict__ W,
    float* __restrict__ out, int N, int K) {
  __shared__ __align__(16) unsigned int Asb[128 * 16];
  __shared__ __align__(16) unsigned int Bsb[64 * 16];
  const int tid = threadIdx.x;
  const int m0 = blockIdx.y * 128;
  const int n0 = blockIdx.x * 64;
  const int lane = tid & 31, wave = tid >> 5;
  const int wm = (wave & 3) * 32, wn = (wave >> 2) * 32;
  const int l16 = lane & 15, lh = lane >> 4;

  v8f acc[2][2] = {};
  for (int k0 = 0; k0 < K; k0 += 32) {
    if (k0 + 32 < K) {
      if (tid < 128) __builtin_prefetch(A + (size_t)(m0 + tid) * K + k0 + 32, 0, 1);
      else if (tid < 192) __builtin_prefetch(W + (size_t)(n0 + (tid - 128)) * K + k0 + 32, 0, 1);
    }
    for (int i = tid; i < 1024; i += 256) {
      int r = i >> 3, c = i & 7;
      float4 f = *(const float4*)(A + (size_t)(m0 + r) * K + k0 + c * 4);
      Asb[r * 16 + c * 2]     = pack2bf(f.x, f.y);
      Asb[r * 16 + c * 2 + 1] = pack2bf(f.z, f.w);
    }
    for (int i = tid; i < 512; i += 256) {
      int r = i >> 3, c = i & 7;
      float4 f = *(const float4*)(W + (size_t)(n0 + r) * K + k0 + c * 4);
      Bsb[r * 16 + c * 2]     = pack2bf(f.x, f.y);
      Bsb[r * 16 + c * 2 + 1] = pack2bf(f.z, f.w);
    }
    __syncthreads();
    const int4* A4 = (const int4*)Asb;
    const int4* B4 = (const int4*)Bsb;
    B32i4 afr[2], bfr[2];
    #pragma unroll
    for (int mi = 0; mi < 2; ++mi) {
      int r = wm + mi * 16 + l16;
      afr[mi].a = A4[r * 4 + lh];
      afr[mi].b = A4[r * 4 + 2 + lh];
    }
    #pragma unroll
    for (int ni = 0; ni < 2; ++ni) {
      int n = wn + ni * 16 + l16;
      bfr[ni].a = B4[n * 4 + lh];
      bfr[ni].b = B4[n * 4 + 2 + lh];
    }
    #pragma unroll
    for (int mi = 0; mi < 2; ++mi) {
      v16bf a = __builtin_bit_cast(v16bf, afr[mi]);
      #pragma unroll
      for (int ni = 0; ni < 2; ++ni) {
        v16bf b = __builtin_bit_cast(v16bf, bfr[ni]);
        acc[mi][ni] = __builtin_amdgcn_wmma_f32_16x16x32_bf16(
            false, a, false, b, (short)0, acc[mi][ni], false, false);
      }
    }
    __syncthreads();
  }
  #pragma unroll
  for (int mi = 0; mi < 2; ++mi)
    #pragma unroll
    for (int ni = 0; ni < 2; ++ni) {
      int col = n0 + wn + ni * 16 + l16;
      #pragma unroll
      for (int j = 0; j < 8; ++j) {
        int row = m0 + wm + mi * 16 + lh * 8 + j;
        out[(size_t)row * N + col] = acc[mi][ni][j];
      }
    }
}

__global__ void __launch_bounds__(64) k_attn(
    const float* __restrict__ q, const float* __restrict__ k,
    const float* __restrict__ v, float* __restrict__ y) {
  __shared__ float Qs[64][65];
  __shared__ float Ks[32][64];
  __shared__ float Vs[32][64];
  const int b = blockIdx.z, h = blockIdx.y, qt0 = blockIdx.x * 64;
  const int tid = threadIdx.x;
  for (int i = tid; i < 64 * 16; i += 64) {
    int r = i >> 4, c = (i & 15) * 4;
    float4 f = *(const float4*)(q + (size_t)(b * TT + qt0 + r) * NEMBD + h * HD + c);
    Qs[r][c] = f.x; Qs[r][c+1] = f.y; Qs[r][c+2] = f.z; Qs[r][c+3] = f.w;
  }
  float acc[64];
  #pragma unroll
  for (int d = 0; d < 64; ++d) acc[d] = 0.f;
  float mrun = -1e30f, lrun = 0.f;
  const int r = tid, qg = qt0 + r;
  for (int j0 = 0; j0 < qt0 + 64; j0 += 32) {
    __syncthreads();
    for (int i = tid; i < 32 * 16; i += 64) {
      int rr = i >> 4, c = (i & 15) * 4;
      *(float4*)&Ks[rr][c] = *(const float4*)(k + (size_t)(b * TT + j0 + rr) * NEMBD + h * HD + c);
      *(float4*)&Vs[rr][c] = *(const float4*)(v + (size_t)(b * TT + j0 + rr) * NEMBD + h * HD + c);
    }
    __syncthreads();
    for (int jj = 0; jj < 32; ++jj) {
      int kg = j0 + jj;
      if (kg > qg) break;
      float sdot = 0.f;
      #pragma unroll
      for (int d = 0; d < 64; ++d) sdot += Qs[r][d] * Ks[jj][d];
      sdot *= 0.125f;
      float mn = fmaxf(mrun, sdot);
      float corr = __expf(mrun - mn);
      float p = __expf(sdot - mn);
      lrun = lrun * corr + p;
      mrun = mn;
      #pragma unroll
      for (int d = 0; d < 64; ++d) acc[d] = acc[d] * corr + p * Vs[jj][d];
    }
  }
  float inv = 1.f / lrun;
  for (int c = 0; c < 64; c += 4) {
    float4 o = {acc[c]*inv, acc[c+1]*inv, acc[c+2]*inv, acc[c+3]*inv};
    *(float4*)(y + (size_t)(b * TT + qg) * NEMBD + h * HD + c) = o;
  }
}

extern "C" void kernel_launch(void* const* d_in, const int* in_sizes, int n_in,
                              void* d_out, int out_size, void* d_ws, size_t ws_size,
                              hipStream_t stream) {
  (void)in_sizes; (void)n_in; (void)out_size; (void)ws_size;
  const int*   idx  = (const int*)  d_in[0];
  const float* tok  = (const float*)d_in[1];
  const float* pos  = (const float*)d_in[2];
  const float* rms1 = (const float*)d_in[3];
  const float* rms2 = (const float*)d_in[4];
  const float* wq_n = (const float*)d_in[5];
  const float* wk_n = (const float*)d_in[6];
  const float* wv_n = (const float*)d_in[7];
  const float* wo_n = (const float*)d_in[8];
  const float* w1_n = (const float*)d_in[9];
  const float* w2_n = (const float*)d_in[10];
  const float* wt[6] = {(const float*)d_in[11], (const float*)d_in[12], (const float*)d_in[13],
                        (const float*)d_in[14], (const float*)d_in[15], (const float*)d_in[16]};
  const float* fnw  = (const float*)d_in[17];
  const float* lmw  = (const float*)d_in[18];
  float* logits = (float*)d_out;

  char* ws = (char*)d_ws;
  size_t off = 0;
  auto alloc = [&](size_t bytes) -> void* {
    void* p = ws + off;
    off = (off + bytes + 255) & ~(size_t)255;
    return p;
  };
  int8_t* qwbuf = (int8_t*)alloc((size_t)NLAYER * 12u * 1024 * 1024);
  float*  wsc   = (float*)alloc(24 * sizeof(float));
  float*  part  = (float*)alloc(128 * sizeof(float));
  float*  x     = (float*)alloc((size_t)MM * NEMBD * 4);
  float*  xa    = (float*)alloc((size_t)MM * NEMBD * 4);
  int8_t* qact  = (int8_t*)alloc((size_t)MM * FFND);
  float*  asc   = (float*)alloc((size_t)MM * 4);
  float*  qb    = (float*)alloc((size_t)MM * NEMBD * 4);
  float*  kb    = (float*)alloc((size_t)MM * NEMBD * 4);
  float*  vb    = (float*)alloc((size_t)MM * NEMBD * 4);
  float*  yb    = (float*)alloc((size_t)MM * NEMBD * 4);
  float*  hb    = (float*)alloc((size_t)MM * FFND * 4);

  const size_t wn[6]   = {NEMBD*NEMBD, NEMBD*NEMBD, NEMBD*NEMBD, NEMBD*NEMBD,
                          (size_t)FFND*NEMBD, (size_t)NEMBD*FFND};
  const size_t qoff[6] = {0, 1u<<20, 2u<<20, 3u<<20, 4u<<20, 8u<<20};

  for (int li = 0; li < NLAYER; ++li)
    for (int t = 0; t < 6; ++t) {
      const float* wp = wt[t] + (size_t)li * wn[t];
      int n = (int)wn[t];
      int8_t* qp = qwbuf + (size_t)li * (12u << 20) + qoff[t];
      float* sp = wsc + li * 6 + t;
      k_absmean_part<<<128, 256, 0, stream>>>(wp, part, n);
      k_absmean_fin<<<1, 128, 0, stream>>>(part, sp, 1.0f / (float)n);
      k_wquant<<<256, 256, 0, stream>>>(wp, sp, qp, n);
    }

  k_embed<<<MM, 256, 0, stream>>>(idx, tok, pos, x);

  dim3 gC(NEMBD / 64, MM / 128);
  dim3 gF(FFND / 64, MM / 128);
  for (int li = 0; li < NLAYER; ++li) {
    int8_t* qL = qwbuf + (size_t)li * (12u << 20);
    float* scL = wsc + li * 6;
    k_rmsnorm<<<MM, 256, 0, stream>>>(x, rms1 + li * NEMBD, xa, NEMBD);
    k_rq<<<MM, 256, 0, stream>>>(xa, wq_n + li * NEMBD, qact, asc, NEMBD);
    k_gemm_i8<<<gC, 256, 0, stream>>>(qact, asc, qL + qoff[0], scL + 0, nullptr, qb, NEMBD, NEMBD, 0);
    k_rq<<<MM, 256, 0, stream>>>(xa, wk_n + li * NEMBD, qact, asc, NEMBD);
    k_gemm_i8<<<gC, 256, 0, stream>>>(qact, asc, qL + qoff[1], scL + 1, nullptr, kb, NEMBD, NEMBD, 0);
    k_rq<<<MM, 256, 0, stream>>>(xa, wv_n + li * NEMBD, qact, asc, NEMBD);
    k_gemm_i8<<<gC, 256, 0, stream>>>(qact, asc, qL + qoff[2], scL + 2, nullptr, vb, NEMBD, NEMBD, 0);
    k_attn<<<dim3(TT / 64, NHEAD, BB), 64, 0, stream>>>(qb, kb, vb, yb);
    k_rq<<<MM, 256, 0, stream>>>(yb, wo_n + li * NEMBD, qact, asc, NEMBD);
    k_gemm_i8<<<gC, 256, 0, stream>>>(qact, asc, qL + qoff[3], scL + 3, x, x, NEMBD, NEMBD, 0);
    k_rmsnorm<<<MM, 256, 0, stream>>>(x, rms2 + li * NEMBD, xa, NEMBD);
    k_rq<<<MM, 256, 0, stream>>>(xa, w1_n + li * NEMBD, qact, asc, NEMBD);
    k_gemm_i8<<<gF, 256, 0, stream>>>(qact, asc, qL + qoff[4], scL + 4, nullptr, hb, FFND, NEMBD, 1);
    k_rq<<<MM, 256, 0, stream>>>(hb, w2_n + li * FFND, qact, asc, FFND);
    k_gemm_i8<<<gC, 256, 0, stream>>>(qact, asc, qL + qoff[5], scL + 5, x, x, NEMBD, FFND, 0);
  }

  k_rmsnorm<<<MM, 256, 0, stream>>>(x, fnw, xa, NEMBD);
  k_gemm_bf16<<<dim3(VOCAB / 64, MM / 128), 256, 0, stream>>>(xa, lmw, logits, VOCAB, NEMBD);
}
